// LSTM_24584392802754
// MI455X (gfx1250) — compile-verified, matrix operand order checked by reading, not run on hardware
//
#include <hip/hip_runtime.h>

// CDNA5 / gfx1250 LSTM+head. Wave32, WMMA f32_16x16x32_f16.
// One wave per 16-batch tile; the whole per-step projection
// (x@W_ih^T + h@W_hh^T + b_ih + b_hh) is fused into 4 WMMAs via an
// augmented K=32 contraction: K0..15 = W_hh^T, K16..20 = W_ih^T,
// K21 = bias row (A supplies 1.0), K22..31 = zero pad.

typedef __attribute__((ext_vector_type(16))) _Float16 v16h;
typedef __attribute__((ext_vector_type(8)))  float    v8f;

#define B_TOT 256
#define T_TOT 8192
#define N_IN  5
#define HID   16

__device__ __forceinline__ float fast_rcp(float x) { return __builtin_amdgcn_rcpf(x); }
__device__ __forceinline__ float sig_f(float x)    { return fast_rcp(1.0f + __expf(-x)); }
__device__ __forceinline__ float tanh_f(float x) {
  float e = __expf(2.0f * x);
  return (e - 1.0f) * fast_rcp(e + 1.0f);
}

__global__ __launch_bounds__(32)
void lstm_head_wmma(const float* __restrict__ x,
                    const float* __restrict__ W_ih,
                    const float* __restrict__ W_hh,
                    const float* __restrict__ b_ih,
                    const float* __restrict__ b_hh,
                    const float* __restrict__ fc_w,
                    const float* __restrict__ fc_b,
                    float* __restrict__ out)
{
  __shared__ _Float16 hlds[16 * 16];          // 16x16 h tile for D->A transpose

  const int lane  = threadIdx.x;              // 0..31
  const int m     = lane & 15;                // row / column within tile
  const int hi    = lane >> 4;                // lane half
  const int kbase = 8 * hi;                   // 16-bit operand K base for this half
  const int tile  = blockIdx.x;               // 16 tiles of 16 batches

  // ---------------- constant B operands: 4 gate tiles, 32x16 f16 each -----
  // element j of v16h <-> K = (j<8 ? kbase+j : 16+kbase+(j-8)), N = 16g+m
  v16h Bt[4];
  #pragma unroll
  for (int g = 0; g < 4; ++g) {
    const int n = g * 16 + m;                 // gate column 0..63 (i,f,g,o groups)
    #pragma unroll
    for (int j = 0; j < 16; ++j) {
      const int k = (j < 8) ? (kbase + j) : (16 + kbase + (j - 8));
      float v;
      if      (k < 16)  v = W_hh[n * HID + k];        // recurrent weights
      else if (k < 21)  v = W_ih[n * N_IN + (k - 16)];// input weights
      else if (k == 21) v = b_ih[n] + b_hh[n];        // fused bias row
      else              v = 0.0f;                     // pad
      Bt[g][j] = (_Float16)v;
    }
  }

  // per-lane head weights matching A-layout h (lane holds h[m][kbase..kbase+7])
  float wl[8];
  #pragma unroll
  for (int i = 0; i < 8; ++i) wl[i] = fc_w[kbase + i];
  const float fcb = fc_b[0];

  // ---------------- A operand ---------------------------------------------
  // elements 0..7  : h part  (K = kbase..kbase+7)
  // elements 8..15 : lanes 0-15 -> K=16..23 = {x0..x4, 1.0(bias), 0, 0}
  //                  lanes 16-31 -> K=24..31 = 0
  v16h A = {};
  if (hi == 0) A[13] = (_Float16)1.0f;        // bias multiplier (K=21)

  float cst[8];
  #pragma unroll
  for (int r = 0; r < 8; ++r) cst[r] = 0.0f;  // c0 = 0 (h0 = 0 via A zero-init)

  const float* xb   = x   + (size_t)(tile * 16 + m) * T_TOT * N_IN;
  float*       outb = out + (size_t)(tile * 16 + m) * T_TOT;

  for (int t = 0; t < T_TOT; ++t) {
    // ---- stream x_t into the A x-slots (lanes 0-15 only, read-once -> NT)
    if (hi == 0) {
      const float* xp = xb + t * N_IN;
      A[8]  = (_Float16)__builtin_nontemporal_load(xp + 0);
      A[9]  = (_Float16)__builtin_nontemporal_load(xp + 1);
      A[10] = (_Float16)__builtin_nontemporal_load(xp + 2);
      A[11] = (_Float16)__builtin_nontemporal_load(xp + 3);
      A[12] = (_Float16)__builtin_nontemporal_load(xp + 4);
    }

    // ---- 4 WMMAs: gates[16 batch x 64] in one K=32 contraction, C = 0
    v8f zc = {};
    v8f gi = __builtin_amdgcn_wmma_f32_16x16x32_f16(false, A, false, Bt[0], (short)0, zc, false, false);
    v8f gf = __builtin_amdgcn_wmma_f32_16x16x32_f16(false, A, false, Bt[1], (short)0, zc, false, false);
    v8f gg = __builtin_amdgcn_wmma_f32_16x16x32_f16(false, A, false, Bt[2], (short)0, zc, false, false);
    v8f go = __builtin_amdgcn_wmma_f32_16x16x32_f16(false, A, false, Bt[3], (short)0, zc, false, false);

    // ---- LSTM cell elementwise; D layout: reg r -> (b = r+8*hi, j = m)
    _Float16 hh[8];
    #pragma unroll
    for (int r = 0; r < 8; ++r) {
      float iv = sig_f(gi[r]);
      float fv = sig_f(gf[r]);
      float gv = tanh_f(gg[r]);
      float ov = sig_f(go[r]);
      float c  = fv * cst[r] + iv * gv;
      cst[r]   = c;
      hh[r]    = (_Float16)(ov * tanh_f(c));
    }

    // ---- D -> A transpose of h through LDS (single wave: LDS is in-order)
    #pragma unroll
    for (int r = 0; r < 8; ++r)
      hlds[(r + 8 * hi) * 16 + m] = hh[r];

    __asm volatile("s_wait_dscnt 0" ::: "memory");  // order scatter vs gather

    #pragma unroll
    for (int j = 0; j < 8; ++j)
      A[j] = hlds[m * 16 + kbase + j];              // contiguous per-lane gather

    // ---- head: out[b,t] = h_t . fc_w + fc_b  (pair halves via xor-16)
    float part = 0.0f;
    #pragma unroll
    for (int i = 0; i < 8; ++i) part += (float)A[i] * wl[i];
    part += __shfl_xor(part, 16, 32);
    if (hi == 0)
      __builtin_nontemporal_store(part + fcb, outb + t);
  }
}

extern "C" void kernel_launch(void* const* d_in, const int* in_sizes, int n_in,
                              void* d_out, int out_size, void* d_ws, size_t ws_size,
                              hipStream_t stream) {
  (void)in_sizes; (void)n_in; (void)d_ws; (void)ws_size; (void)out_size;
  const float* x    = (const float*)d_in[0];
  const float* W_ih = (const float*)d_in[1];
  const float* W_hh = (const float*)d_in[2];
  const float* b_ih = (const float*)d_in[3];
  const float* b_hh = (const float*)d_in[4];
  const float* fc_w = (const float*)d_in[5];
  const float* fc_b = (const float*)d_in[6];
  float* out = (float*)d_out;

  dim3 grid(B_TOT / 16);   // 16 batch tiles
  dim3 block(32);          // one wave32 per tile
  lstm_head_wmma<<<grid, block, 0, stream>>>(x, W_ih, W_hh, b_ih, b_hh, fc_w, fc_b, out);
}
